// JointSelfAttentionLayer_16561393893951
// MI455X (gfx1250) — matrix operand order repaired and checked by reading, not compiled, not run on hardware
//
#include <hip/hip_runtime.h>

// Problem constants (from reference): B=8, SC=SD=2048, D=1024
#define BB     8
#define SEQ    2048
#define D_DIM  1024

typedef __attribute__((ext_vector_type(16))) __bf16 v16bf;
typedef __attribute__((ext_vector_type(8)))  __bf16 v8bf;
typedef __attribute__((ext_vector_type(8)))  float  v8f;

__device__ __forceinline__ unsigned short f32_to_bf16(float f) {
    unsigned int x = __float_as_uint(f);
    unsigned int r = x + 0x7FFFu + ((x >> 16) & 1u);   // round-to-nearest-even
    return (unsigned short)(r >> 16);
}

__device__ __forceinline__ v8f v8f_zero() {
    v8f v;
#pragma unroll
    for (int i = 0; i < 8; ++i) v[i] = 0.0f;
    return v;
}

// A fragment (16x32 bf16, M x K): lane<16 holds row M=lane, K chunks {0..7,16..23};
// lane>=16 holds row M=lane-16, K chunks {8..15,24..31}. Source: row-major, given elem stride.
__device__ __forceinline__ v16bf load_frag_a16(const unsigned short* base, int stride, int lane) {
    int r = lane & 15, half = lane >> 4;
    const unsigned short* p = base + (long)r * stride + half * 8;
    union { v16bf v; v8bf h[2]; } u;
    u.h[0] = *(const v8bf*)(p);
    u.h[1] = *(const v8bf*)(p + 16);
    return u.v;
}

// B fragment (32x16 bf16, K x N): lane holds column N=lane&15, K chunks {0..7,16..23} for
// lane<16, {8..15,24..31} for lane>=16. Source: N-major (i.e. B^T row-major), given elem stride.
__device__ __forceinline__ v16bf load_frag_b16(const unsigned short* base, int stride, int lane) {
    int n = lane & 15, half = lane >> 4;
    const unsigned short* p = base + (long)n * stride + half * 8;
    union { v16bf v; v8bf h[2]; } u;
    u.h[0] = *(const v8bf*)(p);
    u.h[1] = *(const v8bf*)(p + 16);
    return u.v;
}

__device__ __forceinline__ v8f wmma_bf16(v16bf a, v16bf b, v8f c) {
    return __builtin_amdgcn_wmma_f32_16x16x32_bf16(false, a, false, b, (short)0, c, false, false);
}

// ---------------------------------------------------------------------------
// Kernel 1: projections  Q = bf16(x_c @ W_qc),  K = bf16(x_d @ W_kd)
// WG tile 128(M) x 64(N), 8 waves in 4x2, each wave 32x32 (2x2 WMMA tiles).
// ---------------------------------------------------------------------------
__global__ __launch_bounds__(256) void proj_kernel(
    const float* __restrict__ Xc, const float* __restrict__ Xd,
    const float* __restrict__ Wq, const float* __restrict__ Wk,
    unsigned short* __restrict__ Qbf, unsigned short* __restrict__ Kbf) {

    const float* X = (blockIdx.z == 0) ? Xc : Xd;
    const float* W = (blockIdx.z == 0) ? Wq : Wk;
    unsigned short* O = (blockIdx.z == 0) ? Qbf : Kbf;

    __shared__ __align__(16) unsigned short ldsA[128 * 32]; // 128 rows x 32 k
    __shared__ __align__(16) unsigned short ldsB[64 * 32];  // 64 cols (n-major) x 32 k

    const int tid  = threadIdx.x;
    const int lane = tid & 31;
    const int w    = tid >> 5;
    const int mw   = w >> 1;            // 0..3
    const int nw   = w & 1;             // 0..1
    const long rowBase = (long)blockIdx.x * 128;
    const int  n0      = blockIdx.y * 64;

    v8f acc[2][2];
#pragma unroll
    for (int i = 0; i < 2; ++i)
#pragma unroll
        for (int j = 0; j < 2; ++j) acc[i][j] = v8f_zero();

    for (int kc = 0; kc < D_DIM / 32; ++kc) {
        __syncthreads();
        {   // stage A: 128x32 f32 -> bf16 (2 threads per row, 16 elems each)
            int row = tid >> 1;
            int kh  = (tid & 1) * 16;
            const float* src = X + (rowBase + row) * (long)D_DIM + kc * 32 + kh;
            unsigned short* dst = ldsA + row * 32 + kh;
#pragma unroll
            for (int j = 0; j < 16; ++j) dst[j] = f32_to_bf16(src[j]);
        }
        {   // stage B transposed: ldsB[n][k] = W[k0+k][n0+n]
            int k = tid >> 3;           // 0..31
            int n = (tid & 7) * 8;      // 0..56
            const float* src = W + (long)(kc * 32 + k) * D_DIM + n0 + n;
#pragma unroll
            for (int j = 0; j < 8; ++j) ldsB[(n + j) * 32 + k] = f32_to_bf16(src[j]);
        }
        __syncthreads();
#pragma unroll
        for (int i = 0; i < 2; ++i) {
            v16bf a = load_frag_a16(ldsA + (mw * 32 + i * 16) * 32, 32, lane);
#pragma unroll
            for (int j = 0; j < 2; ++j) {
                v16bf b = load_frag_b16(ldsB + (nw * 32 + j * 16) * 32, 32, lane);
                acc[i][j] = wmma_bf16(a, b, acc[i][j]);
            }
        }
    }

    const int nlo = lane & 15, half = lane >> 4;
#pragma unroll
    for (int i = 0; i < 2; ++i)
#pragma unroll
        for (int j = 0; j < 2; ++j) {
            int mbase = mw * 32 + i * 16 + half * 8;
            int nn    = n0 + nw * 32 + j * 16 + nlo;
#pragma unroll
            for (int r = 0; r < 8; ++r)
                O[(rowBase + mbase + r) * (long)D_DIM + nn] = f32_to_bf16(acc[i][j][r]);
        }
}

// ---------------------------------------------------------------------------
// Kernel 2: colw[b,t] = sum_s softmax_row_s(Q K^T)[t]
// One WG per 16-row block (grid 128 x B = 1024 WGs, 8192 waves). The 8 waves
// split the 2048 columns (16 n-tiles each, processed 4 at a time sharing one
// A fragment). Pass 1 computes per-wave partial row max/denom in registers;
// waves merge stats through LDS. Pass 2 recomputes logits and accumulates
// exp(l-m)/d into an LDS column accumulator (ds_add_f32), then flushes with
// global f32 atomics.
// ---------------------------------------------------------------------------
__global__ __launch_bounds__(256) void attn_colsum_kernel(
    const unsigned short* __restrict__ Qbf,
    const unsigned short* __restrict__ Kbf,
    float* __restrict__ colw) {

    __shared__ __align__(16) unsigned short qtile[16 * D_DIM]; // 32 KB
    __shared__ float colw_lds[SEQ];                            // 8 KB
    __shared__ float lds_m[8][16];
    __shared__ float lds_d[8][16];

    const int tid  = threadIdx.x;
    const int lane = tid & 31;
    const int w    = tid >> 5;
    const int b    = blockIdx.y;
    const int nlo  = lane & 15;
    const int half = lane >> 4;

    const long qRow0 = (long)b * SEQ + blockIdx.x * 16;
    const unsigned short* Qsrc  = Qbf + qRow0 * D_DIM;           // 16 consecutive rows
    const unsigned short* Krows = Kbf + (long)b * SEQ * D_DIM;

    {   // stage Q tile (contiguous 32 KB) as uint4 copies
        const uint4* src = (const uint4*)Qsrc;
        uint4* dst = (uint4*)qtile;
        for (int i = tid; i < 16 * D_DIM / 8; i += 256) dst[i] = src[i];
    }
    for (int t = tid; t < SEQ; t += 256) colw_lds[t] = 0.0f;
    __syncthreads();

    // Per-lane stats for rows M = r + 8*half (broadcast across each 16-lane group).
    float m[8], dsum[8];
#pragma unroll
    for (int r = 0; r < 8; ++r) { m[r] = -3.0e38f; dsum[r] = 0.0f; }

    // ---- pass 1: partial (per-wave) online max + denom over this wave's columns ----
    for (int g = 0; g < 4; ++g) {
        const int nt0 = w * 16 + g * 4;                   // 4 n-tiles per group
        const unsigned short* B0 = Krows + (long)nt0 * 16 * D_DIM;
        if (g < 3)   // prefetch next group's K rows (global_prefetch_b8)
            __builtin_prefetch(B0 + (long)(64 + lane * 2) * D_DIM, 0, 1);
        v8f acc[4];
#pragma unroll
        for (int j = 0; j < 4; ++j) acc[j] = v8f_zero();
        for (int kc = 0; kc < D_DIM / 32; ++kc) {
            v16bf a = load_frag_a16(qtile + kc * 32, D_DIM, lane);
#pragma unroll
            for (int j = 0; j < 4; ++j) {
                v16bf bb = load_frag_b16(B0 + (long)j * 16 * D_DIM + kc * 32, D_DIM, lane);
                acc[j] = wmma_bf16(a, bb, acc[j]);
            }
        }
#pragma unroll
        for (int j = 0; j < 4; ++j) {
#pragma unroll
            for (int r = 0; r < 8; ++r) {
                float v = acc[j][r];
                float tmax = v;
                for (int mask = 1; mask < 16; mask <<= 1)
                    tmax = fmaxf(tmax, __shfl_xor(tmax, mask, 16));
                float mnew = fmaxf(m[r], tmax);
                float e = __expf(v - mnew);
                float esum = e;
                for (int mask = 1; mask < 16; mask <<= 1)
                    esum += __shfl_xor(esum, mask, 16);
                dsum[r] = dsum[r] * __expf(m[r] - mnew) + esum;
                m[r] = mnew;
            }
        }
    }

    // publish per-wave stats (values identical across each 16-lane group)
    if (nlo == 0) {
#pragma unroll
        for (int r = 0; r < 8; ++r) {
            lds_m[w][r + 8 * half] = m[r];
            lds_d[w][r + 8 * half] = dsum[r];
        }
    }
    __syncthreads();

    // merge stats across the 8 waves
    float rd[8];
#pragma unroll
    for (int r = 0; r < 8; ++r) {
        const int row = r + 8 * half;
        float mstar = lds_m[0][row];
#pragma unroll
        for (int ww = 1; ww < 8; ++ww) mstar = fmaxf(mstar, lds_m[ww][row]);
        float dstar = 0.0f;
#pragma unroll
        for (int ww = 0; ww < 8; ++ww)
            dstar += lds_d[ww][row] * __expf(lds_m[ww][row] - mstar);
        m[r]  = mstar;
        rd[r] = 1.0f / dstar;
    }

    // ---- pass 2: recompute logits, accumulate column sums ----
    for (int g = 0; g < 4; ++g) {
        const int nt0 = w * 16 + g * 4;
        const unsigned short* B0 = Krows + (long)nt0 * 16 * D_DIM;
        v8f acc[4];
#pragma unroll
        for (int j = 0; j < 4; ++j) acc[j] = v8f_zero();
        for (int kc = 0; kc < D_DIM / 32; ++kc) {
            v16bf a = load_frag_a16(qtile + kc * 32, D_DIM, lane);
#pragma unroll
            for (int j = 0; j < 4; ++j) {
                v16bf bb = load_frag_b16(B0 + (long)j * 16 * D_DIM + kc * 32, D_DIM, lane);
                acc[j] = wmma_bf16(a, bb, acc[j]);
            }
        }
#pragma unroll
        for (int j = 0; j < 4; ++j) {
            float psum = 0.0f;
#pragma unroll
            for (int r = 0; r < 8; ++r) psum += __expf(acc[j][r] - m[r]) * rd[r];
            atomicAdd(&colw_lds[(nt0 + j) * 16 + nlo], psum);   // ds_add_f32
        }
    }
    __syncthreads();
    for (int t = tid; t < SEQ; t += 256)
        atomicAdd(&colw[(long)b * SEQ + t], colw_lds[t]);
}

// ---------------------------------------------------------------------------
// Kernel 3: dst[b,e] = sum_t wgt[b,t] * src[b,t,e]   (wgt==null -> all ones)
// ---------------------------------------------------------------------------
__global__ __launch_bounds__(256) void colsum_kernel(
    const float* __restrict__ src, const float* __restrict__ wgt,
    float* __restrict__ dst, int S) {
    const int e = blockIdx.x * 256 + threadIdx.x;
    const int b = blockIdx.y;
    const int t0 = blockIdx.z * (S / 8);
    const float* p = src + ((long)b * S + t0) * D_DIM + e;
    float acc = 0.0f;
    if (wgt) {
        const float* wp = wgt + (long)b * S + t0;
        for (int t = 0; t < S / 8; ++t) acc += wp[t] * p[(long)t * D_DIM];
    } else {
        for (int t = 0; t < S / 8; ++t) acc += p[(long)t * D_DIM];
    }
    atomicAdd(&dst[(long)b * D_DIM + e], acc);
}

// ---------------------------------------------------------------------------
// Kernel 4: out[b,e] = scale * sum_k u[b,k] * W[k,e]
// ---------------------------------------------------------------------------
__global__ __launch_bounds__(256) void gemv_kernel(
    const float* __restrict__ u, const float* __restrict__ W,
    float* __restrict__ out, float scale) {
    const int e = blockIdx.x * 256 + threadIdx.x;
    const int b = blockIdx.y;
    const float* ub = u + (long)b * D_DIM;
    float acc = 0.0f;
    for (int k = 0; k < D_DIM; ++k) acc += ub[k] * W[(long)k * D_DIM + e];
    out[(long)b * D_DIM + e] = acc * scale;
}

extern "C" void kernel_launch(void* const* d_in, const int* in_sizes, int n_in,
                              void* d_out, int out_size, void* d_ws, size_t ws_size,
                              hipStream_t stream) {
    (void)in_sizes; (void)n_in; (void)out_size; (void)ws_size;

    const float* x_c  = (const float*)d_in[0];
    const float* x_d  = (const float*)d_in[1];
    const float* W_qc = (const float*)d_in[2];
    const float* W_vc = (const float*)d_in[3];
    const float* W_kd = (const float*)d_in[4];
    const float* W_vd = (const float*)d_in[5];

    float* C_out = (float*)d_out;                      // [B, D]
    float* D_out = C_out + (size_t)BB * D_DIM;         // [B, D]

    char* ws = (char*)d_ws;
    const size_t nQK = (size_t)BB * SEQ * D_DIM;
    unsigned short* Qbf = (unsigned short*)ws;  ws += nQK * 2;    // 32 MB
    unsigned short* Kbf = (unsigned short*)ws;  ws += nQK * 2;    // 32 MB
    float* colw  = (float*)ws;  ws += (size_t)BB * SEQ   * 4;     // 64 KB
    float* sumxc = (float*)ws;  ws += (size_t)BB * D_DIM * 4;     // 32 KB
    float* uacc  = (float*)ws;  ws += (size_t)BB * D_DIM * 4;     // 32 KB

    hipMemsetAsync(colw,  0, (size_t)BB * SEQ   * 4, stream);
    hipMemsetAsync(sumxc, 0, (size_t)BB * D_DIM * 4, stream);
    hipMemsetAsync(uacc,  0, (size_t)BB * D_DIM * 4, stream);

    dim3 gProj(BB * SEQ / 128, D_DIM / 64, 2);         // 128 x 16 x 2
    proj_kernel<<<gProj, 256, 0, stream>>>(x_c, x_d, W_qc, W_kd, Qbf, Kbf);

    dim3 gAttn(SEQ / 16, BB, 1);                       // 128 x 8 = 1024 WGs
    attn_colsum_kernel<<<gAttn, 256, 0, stream>>>(Qbf, Kbf, colw);

    dim3 gCol(D_DIM / 256, BB, 8);                     // 4 x 8 x 8
    colsum_kernel<<<gCol, 256, 0, stream>>>(x_c, nullptr, sumxc, SEQ);
    colsum_kernel<<<gCol, 256, 0, stream>>>(x_d, colw,    uacc,  SEQ);

    // A includes 1/sqrt(D)=1/32 applied after softmax; means divide by SC=SD=2048.
    const float scale = 1.0f / (2048.0f * 32.0f);
    dim3 gGemv(D_DIM / 256, BB, 1);
    gemv_kernel<<<gGemv, 256, 0, stream>>>(uacc,  W_vd, C_out, scale);
    gemv_kernel<<<gGemv, 256, 0, stream>>>(sumxc, W_vc, D_out, scale);
}
